// resGATv2Conv_15401752723462
// MI455X (gfx1250) — compile-verified, matrix operand order checked by reading, not run on hardware
//
#include <hip/hip_runtime.h>
#include <math.h>

typedef __attribute__((ext_vector_type(16))) _Float16 v16h;
typedef __attribute__((ext_vector_type(8)))  float    v8f;

#define CH 64      // channels
#define NH 4       // heads
#define ED 16      // edge feature dim
#define HC 256     // NH*CH
#define NEG_SLOPE 0.2f
#define BN_EPS 1e-5f

__device__ __forceinline__ void atomicAddF(float* p, float v) {
  __hip_atomic_fetch_add(p, v, __ATOMIC_RELAXED, __HIP_MEMORY_SCOPE_AGENT);
}
__device__ __forceinline__ void atomicMaxF(float* p, float v) {
  // monotone mapping trick: int-max for >=0, uint-min for <0. mx init = -inf.
  if (v >= 0.0f) atomicMax((int*)p, __float_as_int(v));
  else           atomicMin((unsigned int*)p, (unsigned int)__float_as_int(v));
}

__global__ void fill_kernel(float* __restrict__ p, float v, int n) {
  int t = blockIdx.x * blockDim.x + threadIdx.x;
  if (t < n) p[t] = v;
}

// ---------------------------------------------------------------------------
// xl = x @ W_l, xr = x @ W_r via V_WMMA_F32_16X16X32_F16.
// grid.x = ceil(N/16) row tiles, block = 256 (8 waves).
// waves 0..3 -> W_l cols [w*64,w*64+64); waves 4..7 -> W_r same cols.
// A frag (16x32 f16): lane%16 = row M; lane/16 = K-half; vgpr v holds K pair
//   k = (v<4 ? 2v : 16+2(v-4)) + (lane/16)*8
// B frag (32x16 f16): symmetric, lane%16 = col N, same k mapping.
// D (16x16 f32): vgpr m -> row M = m + (lane/16)*8, col N = lane%16.
// Store guard is tile-uniform: EXEC stays full around the WMMAs, and the
// common (full-tile) path is straight-line unpredicated stores.
// ---------------------------------------------------------------------------
__global__ void gemm_xlr_kernel(const float* __restrict__ x,
                                const float* __restrict__ Wl,
                                const float* __restrict__ Wr,
                                float* __restrict__ xl,
                                float* __restrict__ xr, int N) {
  int tile = blockIdx.x;
  int w    = threadIdx.x >> 5;
  int lane = threadIdx.x & 31;
  int r    = lane & 15;
  int hi   = lane >> 4;

  const float* W   = (w & 4) ? Wr : Wl;
  float*       dst = (w & 4) ? xr : xl;
  int colbase = (w & 3) * 64;

  const bool full = (tile * 16 + 16 <= N);   // uniform across the block

  int arow = tile * 16 + r;
  if (arow >= N) arow = N - 1;               // clamp (keeps EXEC full)
  const float* xrow = x + (size_t)arow * CH;

  v16h a0, a1;
#pragma unroll
  for (int v = 0; v < 8; ++v) {
    int k = ((v < 4) ? 2 * v : 16 + 2 * (v - 4)) + hi * 8;
    a0[2 * v]     = (_Float16)xrow[k];
    a0[2 * v + 1] = (_Float16)xrow[k + 1];
    a1[2 * v]     = (_Float16)xrow[32 + k];
    a1[2 * v + 1] = (_Float16)xrow[32 + k + 1];
  }

#pragma unroll
  for (int ct = 0; ct < 4; ++ct) {
    int col = colbase + ct * 16 + r;
    v16h b0, b1;
#pragma unroll
    for (int v = 0; v < 8; ++v) {
      int k = ((v < 4) ? 2 * v : 16 + 2 * (v - 4)) + hi * 8;
      b0[2 * v]     = (_Float16)W[(size_t)k * HC + col];
      b0[2 * v + 1] = (_Float16)W[(size_t)(k + 1) * HC + col];
      b1[2 * v]     = (_Float16)W[(size_t)(32 + k) * HC + col];
      b1[2 * v + 1] = (_Float16)W[(size_t)(33 + k) * HC + col];
    }
    v8f c = {};
    c = __builtin_amdgcn_wmma_f32_16x16x32_f16(false, a0, false, b0,
                                               (short)0, c, false, false);
    c = __builtin_amdgcn_wmma_f32_16x16x32_f16(false, a1, false, b1,
                                               (short)0, c, false, false);
    if (full) {
#pragma unroll
      for (int m = 0; m < 8; ++m) {
        int row = tile * 16 + m + hi * 8;
        dst[(size_t)row * HC + col] = c[m];
      }
    } else {
#pragma unroll
      for (int m = 0; m < 8; ++m) {
        int row = tile * 16 + m + hi * 8;
        if (row < N) dst[(size_t)row * HC + col] = c[m];
      }
    }
  }
}

// --- self-loop attr: segment sum + count over dst --------------------------
__global__ void loop_sum_kernel(const int* __restrict__ dst,
                                const float* __restrict__ ea,
                                float* __restrict__ lsum,
                                float* __restrict__ cnt, int E) {
  int e = blockIdx.x * blockDim.x + threadIdx.x;
  if (e >= E) return;
  int d = dst[e];
  atomicAddF(cnt + d, 1.0f);
  const float* a = ea + (size_t)e * ED;
  float* s = lsum + (size_t)d * ED;
#pragma unroll
  for (int k = 0; k < ED; ++k) atomicAddF(s + k, a[k]);
}

__global__ void loop_div_kernel(float* __restrict__ lsum,
                                const float* __restrict__ cnt, int N) {
  int t = blockIdx.x * blockDim.x + threadIdx.x;
  if (t >= N * ED) return;
  float c = cnt[t >> 4];
  lsum[t] /= (c < 1.0f ? 1.0f : c);
}

// --- pass 1: attention logits + segment max --------------------------------
__global__ void attn_logits_kernel(const int* __restrict__ src,
                                   const int* __restrict__ dst,
                                   const float* __restrict__ ea,
                                   const float* __restrict__ lattr,
                                   const float* __restrict__ We,
                                   const float* __restrict__ att,
                                   const float* __restrict__ xl,
                                   const float* __restrict__ xr,
                                   float* __restrict__ logits,
                                   float* __restrict__ mx, int E, int N) {
  __shared__ float sWe[ED * HC];
  __shared__ float sAtt[NH * CH];
  for (int i = threadIdx.x; i < ED * HC; i += blockDim.x) sWe[i] = We[i];
  for (int i = threadIdx.x; i < NH * CH; i += blockDim.x) sAtt[i] = att[i];
  __syncthreads();

  int t = blockIdx.x * blockDim.x + threadIdx.x;
  if (t >= (E + N) * NH) return;
  int e = t >> 2, h = t & 3;
  int s, d; const float* ap;
  if (e < E) { s = src[e]; d = dst[e]; ap = ea + (size_t)e * ED; }
  else       { s = d = e - E;          ap = lattr + (size_t)(e - E) * ED; }

  float ab[ED];
#pragma unroll
  for (int k = 0; k < ED; ++k) ab[k] = ap[k];

  const float4* xls = (const float4*)(xl + (size_t)s * HC + h * CH);
  const float4* xrd = (const float4*)(xr + (size_t)d * HC + h * CH);
  const float* wj = sWe + h * CH;
  const float* aj = sAtt + h * CH;

  float logit = 0.0f;
  for (int c4 = 0; c4 < CH / 4; ++c4) {
    float4 vl = xls[c4];
    float4 vr = xrd[c4];
    float ms[4] = {vl.x + vr.x, vl.y + vr.y, vl.z + vr.z, vl.w + vr.w};
#pragma unroll
    for (int j = 0; j < 4; ++j) {
      int c = c4 * 4 + j;
      float ee = 0.0f;
#pragma unroll
      for (int k = 0; k < ED; ++k) ee = fmaf(ab[k], wj[k * HC + c], ee);
      float m = ms[j] + ee;
      m = (m > 0.0f) ? m : NEG_SLOPE * m;
      logit = fmaf(m, aj[c], logit);
    }
  }
  logits[t] = logit;
  atomicMaxF(mx + (size_t)d * NH + h, logit);
}

// --- pass 2: exp + segment sum (denominator) -------------------------------
__global__ void softmax_den_kernel(const int* __restrict__ dst,
                                   float* __restrict__ logits,
                                   const float* __restrict__ mx,
                                   float* __restrict__ den, int E, int N) {
  int t = blockIdx.x * blockDim.x + threadIdx.x;
  if (t >= (E + N) * NH) return;
  int e = t >> 2, h = t & 3;
  int d = (e < E) ? dst[e] : (e - E);
  float ex = __expf(logits[t] - mx[(size_t)d * NH + h]);
  logits[t] = ex;                       // reuse storage for exp values
  atomicAddF(den + (size_t)d * NH + h, ex);
}

// --- pass 3: alpha-weighted scatter-add of xl[src] -------------------------
__global__ void aggregate_kernel(const int* __restrict__ src,
                                 const int* __restrict__ dst,
                                 const float* __restrict__ logits,
                                 const float* __restrict__ den,
                                 const float* __restrict__ xl,
                                 float* __restrict__ agg, int E, int N) {
  int t = blockIdx.x * blockDim.x + threadIdx.x;
  if (t >= (E + N) * NH) return;
  int e = t >> 2, h = t & 3;
  int s, d;
  if (e < E) { s = src[e]; d = dst[e]; } else { s = d = e - E; }
  float alpha = logits[t] / den[(size_t)d * NH + h];
  const float4* xls = (const float4*)(xl + (size_t)s * HC + h * CH);
  float* ag = agg + (size_t)d * HC + h * CH;
#pragma unroll 4
  for (int c4 = 0; c4 < CH / 4; ++c4) {
    float4 v = xls[c4];
    atomicAddF(ag + 4 * c4 + 0, v.x * alpha);
    atomicAddF(ag + 4 * c4 + 1, v.y * alpha);
    atomicAddF(ag + 4 * c4 + 2, v.z * alpha);
    atomicAddF(ag + 4 * c4 + 3, v.w * alpha);
  }
}

// --- pass 4: head mean + bias; batch sum / sumsq ---------------------------
__global__ void headmean_stats_kernel(const float* __restrict__ agg,
                                      const float* __restrict__ bias,
                                      float* __restrict__ outbuf,
                                      float* __restrict__ stats, int N) {
  __shared__ float s1[256];
  __shared__ float s2[256];
  int t = blockIdx.x * 256 + threadIdx.x;
  float o = 0.0f;
  if (t < N * CH) {
    int n = t >> 6, c = t & 63;
    const float* a = agg + (size_t)n * HC;
    o = 0.25f * (a[c] + a[CH + c] + a[2 * CH + c] + a[3 * CH + c]) + bias[c];
    outbuf[t] = o;
  }
  s1[threadIdx.x] = o;
  s2[threadIdx.x] = o * o;
  __syncthreads();
  if (threadIdx.x < 64) {   // block base is a multiple of 256 -> c == tid
    float a = s1[threadIdx.x] + s1[threadIdx.x + 64] +
              s1[threadIdx.x + 128] + s1[threadIdx.x + 192];
    float b = s2[threadIdx.x] + s2[threadIdx.x + 64] +
              s2[threadIdx.x + 128] + s2[threadIdx.x + 192];
    atomicAddF(stats + threadIdx.x, a);
    atomicAddF(stats + 64 + threadIdx.x, b);
  }
}

__global__ void bn_finalize_kernel(float* __restrict__ stats, int N) {
  int c = threadIdx.x;
  if (c < CH) {
    float mu  = stats[c] / (float)N;
    float var = stats[64 + c] / (float)N - mu * mu;
    stats[c]      = mu;
    stats[64 + c] = rsqrtf(var + BN_EPS);
  }
}

// --- pass 5: BN + residual + ReLU ------------------------------------------
__global__ void bn_res_relu_kernel(const float* __restrict__ outbuf,
                                   const float* __restrict__ x,
                                   const float* __restrict__ stats,
                                   const float* __restrict__ gamma,
                                   const float* __restrict__ beta,
                                   float* __restrict__ out, int N) {
  int t = blockIdx.x * blockDim.x + threadIdx.x;
  if (t >= N * CH) return;
  int c = t & 63;
  float y = gamma[c] * (outbuf[t] - stats[c]) * stats[64 + c] + beta[c] + x[t];
  out[t] = (y > 0.0f) ? y : 0.0f;
}

extern "C" void kernel_launch(void* const* d_in, const int* in_sizes, int n_in,
                              void* d_out, int out_size, void* d_ws, size_t ws_size,
                              hipStream_t stream) {
  const float* x     = (const float*)d_in[0];
  const int*   ei    = (const int*)d_in[1];
  const float* ea    = (const float*)d_in[2];
  const float* Wl    = (const float*)d_in[3];
  const float* Wr    = (const float*)d_in[4];
  const float* We    = (const float*)d_in[5];
  const float* att   = (const float*)d_in[6];
  const float* bias  = (const float*)d_in[7];
  const float* gamma = (const float*)d_in[8];
  const float* beta  = (const float*)d_in[9];
  float* out = (float*)d_out;

  const int N  = in_sizes[0] / CH;
  const int E  = in_sizes[1] / 2;
  const int EP = E + N;
  const int* src = ei;
  const int* dst = ei + E;

  // workspace layout (floats)
  float* ws     = (float*)d_ws;
  float* xl     = ws;                           // N*256
  float* xr     = xl + (size_t)N * HC;          // N*256
  float* lattr  = xr + (size_t)N * HC;          // N*16   (zero block start)
  float* cnt    = lattr + (size_t)N * ED;       // N
  float* den    = cnt + N;                      // N*4
  float* agg    = den + (size_t)N * NH;         // N*256
  float* stats  = agg + (size_t)N * HC;         // 128    (zero block end)
  float* outbuf = stats + 2 * CH;               // N*64
  float* logits = outbuf + (size_t)N * CH;      // EP*4
  float* mx     = logits + (size_t)EP * NH;     // N*4

  const int T = 256;
  auto nb = [](long n) { return (int)((n + 255) / 256); };

  // init accumulators (every call; graph-replay safe)
  long zcount = (long)N * (ED + 1 + NH + HC) + 2 * CH;
  fill_kernel<<<nb(zcount), T, 0, stream>>>(lattr, 0.0f, (int)zcount);
  fill_kernel<<<nb((long)N * NH), T, 0, stream>>>(mx, -INFINITY, N * NH);

  // dense projections (WMMA)
  gemm_xlr_kernel<<<(N + 15) / 16, 256, 0, stream>>>(x, Wl, Wr, xl, xr, N);

  // self-loop attr mean
  loop_sum_kernel<<<nb(E), T, 0, stream>>>(dst, ea, lattr, cnt, E);
  loop_div_kernel<<<nb((long)N * ED), T, 0, stream>>>(lattr, cnt, N);

  // edge passes
  long tEdge = (long)EP * NH;
  attn_logits_kernel<<<nb(tEdge), T, 0, stream>>>(src, dst, ea, lattr, We, att,
                                                  xl, xr, logits, mx, E, N);
  softmax_den_kernel<<<nb(tEdge), T, 0, stream>>>(dst, logits, mx, den, E, N);
  aggregate_kernel<<<nb(tEdge), T, 0, stream>>>(src, dst, logits, den, xl, agg, E, N);

  // epilogue
  headmean_stats_kernel<<<nb((long)N * CH), 256, 0, stream>>>(agg, bias, outbuf, stats, N);
  bn_finalize_kernel<<<1, 64, 0, stream>>>(stats, N);
  bn_res_relu_kernel<<<nb((long)N * CH), T, 0, stream>>>(outbuf, x, stats, gamma,
                                                         beta, out, N);
}
